// Net_13580686590538
// MI455X (gfx1250) — 8-bit matrix operand order repaired and checked by reading, compiled, not run on hardware
//
#include <hip/hip_runtime.h>

typedef int v8i __attribute__((ext_vector_type(8)));

#define BLK_IMGS 16
#define THREADS  256

__global__ __launch_bounds__(THREADS)
void fused_qnet_wmma(const float* __restrict__ x,
                     const float* __restrict__ conv_w,
                     const float* __restrict__ fc_w,
                     float* __restrict__ out)
{
    __shared__ __align__(16) signed char   s_x[BLK_IMGS][26][28];
    __shared__ __align__(16) unsigned char s_act[BLK_IMGS][256];
    __shared__ __align__(16) signed char   s_w[16][256];

    const int t = threadIdx.x;
    const long long img0 = (long long)blockIdx.x * BLK_IMGS;

    {
        int* p = (int*)&s_x[0][0][0];
        for (int i = t; i < (BLK_IMGS * 26 * 28) / 4; i += THREADS) p[i] = 0;
        int* q = (int*)&s_act[0][0];
        for (int i = t; i < (BLK_IMGS * 256) / 4; i += THREADS) q[i] = 0;
        int* r = (int*)&s_w[0][0];
        for (int i = t; i < (16 * 256) / 4; i += THREADS) r[i] = 0;
    }
    __syncthreads();

    for (int i = t; i < 10 * 196; i += THREADS) {
        int n = i / 196, k = i % 196;
        float w = fc_w[i];
        w = fminf(fmaxf(w, -0.5f), 0.5f);
        s_w[n][k] = (signed char)(int)rintf(w * 2.0f);
    }

    int cw[9];
#pragma unroll
    for (int k = 0; k < 9; ++k) {
        float w = conv_w[k];
        w = fminf(fmaxf(w, -0.5f), 0.5f);
        cw[k] = (int)rintf(w * 2.0f);
    }

    {
        const float4* xv = (const float4*)(x + img0 * 576);
        for (int p4 = t; p4 < BLK_IMGS * 144; p4 += THREADS) {
            int img = p4 / 144;
            int idx = (p4 % 144) * 4;
            int row = idx / 24, col = idx % 24;
            float4 v = xv[p4];
            float f[4] = {v.x, v.y, v.z, v.w};
#pragma unroll
            for (int j = 0; j < 4; ++j) {
                float a = f[j] * (1.0f / 127.5f) - 1.0f;
                a = fminf(fmaxf(a, -127.0f / 128.0f), 127.0f / 128.0f);
                s_x[img][row + 1][col + 1 + j] = (signed char)(int)rintf(a * 128.0f);
            }
        }
    }
    __syncthreads();

    for (int o = t; o < BLK_IMGS * 196; o += THREADS) {
        int img = o / 196;
        int rc  = o % 196;
        int pr = rc / 14, pc = rc % 14;
        int maxv = 0;
#pragma unroll
        for (int dr = 0; dr < 2; ++dr) {
            int rr = 2 * pr - 2 + dr;
            if (rr < 0 || rr > 23) continue;
#pragma unroll
            for (int dc = 0; dc < 2; ++dc) {
                int cc = 2 * pc - 2 + dc;
                if (cc < 0 || cc > 23) continue;
                int s = 0;
#pragma unroll
                for (int kr = 0; kr < 3; ++kr)
#pragma unroll
                    for (int kc = 0; kc < 3; ++kc)
                        s += (int)s_x[img][rr + kr][cc + kc] * cw[kr * 3 + kc];
                maxv = max(maxv, s);
            }
        }
        int a = (int)rintf((float)maxv * 0.5f);
        a = min(a, 127);
        s_act[img][rc] = (unsigned char)a;
    }
    __syncthreads();

    if (t < 32) {
        const int lane = t;
        v8i acc = {0, 0, 0, 0, 0, 0, 0, 0};
#pragma unroll
        for (int st = 0; st < 4; ++st) {
            v8i a, b;
            int rowA = lane & 15;
            int kA   = 64 * st + ((lane >> 4) << 3);
#pragma unroll
            for (int v = 0; v < 8; ++v) {
                int k = kA + ((v & 1) << 2) + (((v >> 1) & 1) << 4) + ((v >> 2) << 5);
                a[v] = *(const int*)&s_act[rowA][k];
            }
            int colB = lane & 15;
            int kB   = 64 * st + ((lane >> 4) << 3);
#pragma unroll
            for (int v = 0; v < 8; ++v) {
                int k = kB + ((v & 1) << 2) + (((v >> 1) & 1) << 4) + ((v >> 2) << 5);
                b[v] = *(const int*)&s_w[colB][k];
            }
            acc = __builtin_amdgcn_wmma_i32_16x16x64_iu8(true, a, true, b, acc,
                                                         false, false);
        }
        int n = lane & 15;
        if (n < 10) {
#pragma unroll
            for (int r = 0; r < 8; ++r) {
                int m = r + ((lane >> 4) << 3);
                out[(img0 + m) * 10 + n] = (float)acc[r] * (1.0f / 2048.0f);
            }
        }
    }
}

extern "C" void kernel_launch(void* const* d_in, const int* in_sizes, int n_in,
                              void* d_out, int out_size, void* d_ws, size_t ws_size,
                              hipStream_t stream) {
    const float* x      = (const float*)d_in[0];
    const float* conv_w = (const float*)d_in[1];
    const float* fc_w   = (const float*)d_in[2];
    float* out = (float*)d_out;

    int B = in_sizes[0] / 576;
    int grid = B / BLK_IMGS;
    fused_qnet_wmma<<<grid, THREADS, 0, stream>>>(x, conv_w, fc_w, out);
}
